// SoAP_20968030339657
// MI455X (gfx1250) — matrix operand order repaired and checked by reading, not compiled, not run on hardware
//
#include <hip/hip_runtime.h>
#include <hip/hip_bf16.h>

// ---------------------------------------------------------------------------
// SoAP: second-order pooling -> matrix log (Newton-Schulz inverse
// scaling-and-squaring, all WMMA) -> power norm -> FC -> L2 norm.
// MI455X / gfx1250, wave32.  All matmuls: v_wmma_f32_16x16x32_bf16 with
// hi/lo bf16 split planes (fp32-class accuracy at bf16 matrix-core rate).
// LDS matrices are bf16 planes: the fp32->bf16 split happens once per store,
// never on fragment loads; symmetry of all NS iterates lets both A and B
// fragments be loaded as contiguous 16B ds_load_b128 reads.
// Power norms use hardware v_log_f32/v_exp_f32 (values pre-clamped > 0).
// ---------------------------------------------------------------------------

typedef __attribute__((ext_vector_type(16))) __bf16 v16bf;
typedef __attribute__((ext_vector_type(8)))  __bf16 v8bf;
typedef __attribute__((ext_vector_type(8)))  float  v8f;

#define EPSV      1e-8f
#define B_BATCH   64
#define NPTS      4096
#define DD        128
#define OUTF      256
#define KFC       (DD * DD)   // 16384
#define NS_LEVELS 4           // 2^4 inverse scaling-and-squaring
#define NS_ITERS  8           // Newton-Schulz iterations per square root

__device__ __forceinline__ v8f wmma_bf16(v16bf a, v16bf b, v8f c) {
  return __builtin_amdgcn_wmma_f32_16x16x32_bf16(false, a, false, b, (short)0, c,
                                                 false, false);
}

__device__ __forceinline__ v16bf cat8(v8bf a, v8bf b) {
  return __builtin_shufflevector(a, b, 0, 1, 2, 3, 4, 5, 6, 7,
                                       8, 9, 10, 11, 12, 13, 14, 15);
}

__device__ __forceinline__ void splitf(float v, __bf16& hi, __bf16& lo) {
  hi = (__bf16)v;
  lo = (__bf16)(v - (float)hi);
}

// fast x^p for x >= EPSV > 0: raw hardware trans ops (v_log_f32 / v_exp_f32)
__device__ __forceinline__ float fast_pow_pos(float x, float p) {
  return __builtin_amdgcn_exp2f(__builtin_amdgcn_logf(x) * p);
}

// ---- fragment loaders (ISA 7.12.2 layouts, wave32) -------------------------
// A (16xK bf16): lane m = mbase+(l&15); element e -> k = (e<8?e:e+8)+8*half.
__device__ __forceinline__ v16bf load_a_plane(const __bf16* p, int stride,
                                              int mbase, int k0) {
  int lane = threadIdx.x & 31;
  int hl   = lane >> 4;
  int m    = mbase + (lane & 15);
  const __bf16* base = p + m * stride + k0 + 8 * hl;
  v8bf r1 = *(const v8bf*)(base);
  v8bf r2 = *(const v8bf*)(base + 16);
  return cat8(r1, r2);
}

// B (Kx16 bf16): lane n = nbase+(l&15); element e -> k = (e<8?e:e+8)+8*half.
// Reads element (k,n) from p[n*stride + k]: symmetric matrices / W^T.
__device__ __forceinline__ v16bf load_b_rowk(const __bf16* p, int stride,
                                             int k0, int nbase) {
  int lane = threadIdx.x & 31;
  int hl   = lane >> 4;
  int n    = nbase + (lane & 15);
  const __bf16* base = p + n * stride + k0 + 8 * hl;
  v8bf r1 = *(const v8bf*)(base);
  v8bf r2 = *(const v8bf*)(base + 16);
  return cat8(r1, r2);
}

__device__ __forceinline__ void zero_acc(v8f* a, int cnt) {
  for (int j = 0; j < cnt; ++j)
#pragma unroll
    for (int r = 0; r < 8; ++r) a[j][r] = 0.0f;
}

// ---- 16x128 strip matmuls over 128x128 bf16-plane LDS matrices (8 waves) ---
__device__ __forceinline__ void mm3(const __bf16* Ah, const __bf16* Al,
                                    const __bf16* Bh, const __bf16* Bl,
                                    v8f acc[8]) {
  int mbase = (threadIdx.x >> 5) * 16;
#pragma unroll
  for (int kc = 0; kc < 4; ++kc) {
    int k0 = kc * 32;
    v16bf ah = load_a_plane(Ah, DD, mbase, k0);
    v16bf al = load_a_plane(Al, DD, mbase, k0);
#pragma unroll
    for (int j = 0; j < 8; ++j) {
      v16bf bh = load_b_rowk(Bh, DD, k0, j * 16);
      v16bf bl = load_b_rowk(Bl, DD, k0, j * 16);
      acc[j] = wmma_bf16(ah, bh, acc[j]);
      acc[j] = wmma_bf16(ah, bl, acc[j]);
      acc[j] = wmma_bf16(al, bh, acc[j]);
    }
  }
}

// A split x B single-plane
__device__ __forceinline__ void mm2_ab(const __bf16* Ah, const __bf16* Al,
                                       const __bf16* Bs, v8f acc[8]) {
  int mbase = (threadIdx.x >> 5) * 16;
#pragma unroll
  for (int kc = 0; kc < 4; ++kc) {
    int k0 = kc * 32;
    v16bf ah = load_a_plane(Ah, DD, mbase, k0);
    v16bf al = load_a_plane(Al, DD, mbase, k0);
#pragma unroll
    for (int j = 0; j < 8; ++j) {
      v16bf bs = load_b_rowk(Bs, DD, k0, j * 16);
      acc[j] = wmma_bf16(ah, bs, acc[j]);
      acc[j] = wmma_bf16(al, bs, acc[j]);
    }
  }
}

// A single-plane x B split
__device__ __forceinline__ void mm2_ba(const __bf16* As,
                                       const __bf16* Bh, const __bf16* Bl,
                                       v8f acc[8]) {
  int mbase = (threadIdx.x >> 5) * 16;
#pragma unroll
  for (int kc = 0; kc < 4; ++kc) {
    int k0 = kc * 32;
    v16bf as = load_a_plane(As, DD, mbase, k0);
#pragma unroll
    for (int j = 0; j < 8; ++j) {
      v16bf bh = load_b_rowk(Bh, DD, k0, j * 16);
      v16bf bl = load_b_rowk(Bl, DD, k0, j * 16);
      acc[j] = wmma_bf16(as, bh, acc[j]);
      acc[j] = wmma_bf16(as, bl, acc[j]);
    }
  }
}

// ---------------------------------------------------------------------------
// Kernel 1: cov[b] = clamp(X)^T clamp(X) / N   (one workgroup per batch).
// ---------------------------------------------------------------------------
__global__ __launch_bounds__(256) void cov_kernel(const float* __restrict__ x,
                                                  float* __restrict__ cov) {
  __shared__ __align__(16) __bf16 cTh[DD * 32];  // 8 KB  chunkT hi plane
  __shared__ __align__(16) __bf16 cTl[DD * 32];  // 8 KB  chunkT lo plane
  int b = blockIdx.x;
  int tid = threadIdx.x;
  int wave = tid >> 5;
  int lane = tid & 31, hl = lane >> 4, n0 = lane & 15;
  const float* xb = x + (size_t)b * NPTS * DD;

  v8f acc[8];
  zero_acc(acc, 8);

  for (int c = 0; c < NPTS / 32; ++c) {
    const float4* src4 = (const float4*)(xb + (size_t)c * 32 * DD);
#pragma unroll 1
    for (int u = tid; u < 32 * DD / 4; u += 256) {
      float4 v = src4[u];
      int f0 = u * 4;
      int nl = f0 >> 7;        // point within chunk (0..31)
      int d  = f0 & (DD - 1);  // dim (0..127), covers d..d+3
      float vv[4] = {v.x, v.y, v.z, v.w};
#pragma unroll
      for (int j = 0; j < 4; ++j) {
        float val = fmaxf(vv[j], EPSV);
        __bf16 h, l;
        splitf(val, h, l);
        cTh[(d + j) * 32 + nl] = h;
        cTl[(d + j) * 32 + nl] = l;
      }
    }
    __syncthreads();

    v16bf ah = load_a_plane(cTh, 32, wave * 16, 0);
    v16bf al = load_a_plane(cTl, 32, wave * 16, 0);
#pragma unroll
    for (int j = 0; j < 8; ++j) {
      v16bf bh = load_b_rowk(cTh, 32, 0, j * 16);
      v16bf bl = load_b_rowk(cTl, 32, 0, j * 16);
      acc[j] = wmma_bf16(ah, bh, acc[j]);
      acc[j] = wmma_bf16(ah, bl, acc[j]);
      acc[j] = wmma_bf16(al, bh, acc[j]);
    }
    __syncthreads();
  }

  const float scale = 1.0f / (float)NPTS;
  float* cb = cov + (size_t)b * DD * DD;
#pragma unroll
  for (int j = 0; j < 8; ++j)
#pragma unroll
    for (int r = 0; r < 8; ++r)
      cb[(wave * 16 + r + 8 * hl) * DD + j * 16 + n0] = acc[j][r] * scale;
}

// ---------------------------------------------------------------------------
// Kernel 2: matrix log via NS square roots + log series, then clip and two
// power-norms.  One workgroup per batch.
// LDS: Yh/Yl/Zh/Zl (bf16, 32 KB each) + T (bf16, 32 KB) = 160 KB.
// ---------------------------------------------------------------------------
__global__ __launch_bounds__(256) void logm_kernel(const float* __restrict__ cov,
                                                   const float* __restrict__ p_ptr,
                                                   __bf16* __restrict__ flatH,
                                                   __bf16* __restrict__ flatL) {
  __shared__ __align__(16) __bf16 Yh[DD * DD];
  __shared__ __align__(16) __bf16 Yl[DD * DD];
  __shared__ __align__(16) __bf16 Zh[DD * DD];
  __shared__ __align__(16) __bf16 Zl[DD * DD];
  __shared__ __align__(16) __bf16 Ts[DD * DD];

  int b = blockIdx.x;
  int tid = threadIdx.x;
  int wave = tid >> 5;
  int lane = tid & 31, hl = lane >> 4, n0 = lane & 15;

  const float* cb = cov + (size_t)b * DD * DD;

  // trace -> spectral normalization  A0 = cov / tr(cov)  (eigs in (0,1]).
  // T plane is unused until the first NS iteration: borrow it as f32 scratch.
  float* scr = (float*)Ts;
  if (tid < DD) scr[tid] = cb[tid * (DD + 1)];
  __syncthreads();
  if (tid == 0) {
    float s = 0.0f;
    for (int i = 0; i < DD; ++i) s += scr[i];
    scr[0] = fmaxf(s, 1e-20f);
  }
  __syncthreads();
  float c    = scr[0];
  float invc = 1.0f / c;
  float logc = logf(c);
  __syncthreads();

#pragma unroll 1
  for (int i = tid; i < DD * DD; i += 256) {
    __bf16 h, l;
    splitf(cb[i] * invc, h, l);
    Yh[i] = h; Yl[i] = l;
  }
  __syncthreads();

  for (int lvl = 0; lvl < NS_LEVELS; ++lvl) {
#pragma unroll 1
    for (int i = tid; i < DD * DD; i += 256) {
      Zh[i] = (__bf16)(((i & (DD - 1)) == (i >> 7)) ? 1.0f : 0.0f);
      Zl[i] = (__bf16)0.0f;
    }
    __syncthreads();

    for (int it = 0; it < NS_ITERS; ++it) {
      // T = 0.5*(3I - Z*Y), transform fused into bf16 store
      v8f acc[8];
      zero_acc(acc, 8);
      mm3(Zh, Zl, Yh, Yl, acc);
#pragma unroll
      for (int j = 0; j < 8; ++j)
#pragma unroll
        for (int r = 0; r < 8; ++r) {
          int m = wave * 16 + r + 8 * hl;
          int n = j * 16 + n0;
          float v = acc[j][r];
          Ts[m * DD + n] = (__bf16)((m == n) ? (1.5f - 0.5f * v) : (-0.5f * v));
        }
      __syncthreads();

      // Z' = T*Z  (reads all Z rows: stage in regs, barrier, then write)
      v8f accz[8];
      zero_acc(accz, 8);
      mm2_ba(Ts, Zh, Zl, accz);
      __syncthreads();
#pragma unroll
      for (int j = 0; j < 8; ++j)
#pragma unroll
        for (int r = 0; r < 8; ++r) {
          int m = wave * 16 + r + 8 * hl;
          int n = j * 16 + n0;
          __bf16 h, l;
          splitf(accz[j][r], h, l);
          Zh[m * DD + n] = h; Zl[m * DD + n] = l;
        }

      // Y' = Y*T  (A rows strip-local; T stable -> safe in-place)
      v8f accy[8];
      zero_acc(accy, 8);
      mm2_ab(Yh, Yl, Ts, accy);
#pragma unroll
      for (int j = 0; j < 8; ++j)
#pragma unroll
        for (int r = 0; r < 8; ++r) {
          int m = wave * 16 + r + 8 * hl;
          int n = j * 16 + n0;
          __bf16 h, l;
          splitf(accy[j][r], h, l);
          Yh[m * DD + n] = h; Yl[m * DD + n] = l;
        }
      __syncthreads();
    }
  }

  // E = Y - I ;  log(A0) ~= 2^s * (E - E^2/2 + E^3/3)
#pragma unroll 1
  for (int i = tid; i < DD * DD; i += 256) {
    float v = (float)Yh[i] + (float)Yl[i] -
              (((i & (DD - 1)) == (i >> 7)) ? 1.0f : 0.0f);
    __bf16 h, l;
    splitf(v, h, l);
    Zh[i] = h; Zl[i] = l;
  }
  __syncthreads();

  v8f acc2[8];
  zero_acc(acc2, 8);
  mm3(Zh, Zl, Zh, Zl, acc2);                    // E^2 (symmetric)
#pragma unroll
  for (int j = 0; j < 8; ++j)
#pragma unroll
    for (int r = 0; r < 8; ++r)
      Ts[(wave * 16 + r + 8 * hl) * DD + j * 16 + n0] = (__bf16)acc2[j][r];
  __syncthreads();

  v8f acc3[8];
  zero_acc(acc3, 8);
  mm2_ab(Zh, Zl, Ts, acc3);                     // E^3

  float p    = p_ptr[0];
  float twos = (float)(1 << NS_LEVELS);
  __bf16* fh = flatH + (size_t)b * DD * DD;
  __bf16* fl = flatL + (size_t)b * DD * DD;
#pragma unroll
  for (int j = 0; j < 8; ++j)
#pragma unroll
    for (int r = 0; r < 8; ++r) {
      int m = wave * 16 + r + 8 * hl;
      int n = j * 16 + n0;
      float e1 = (float)Zh[m * DD + n] + (float)Zl[m * DD + n];
      float v = twos * (e1 - 0.5f * acc2[j][r] + (1.0f / 3.0f) * acc3[j][r]) +
                ((m == n) ? logc : 0.0f);
      v = fmaxf(v, EPSV);                          // clip(m, eps)
      v = fmaxf(fast_pow_pos(v, p), EPSV);         // power-norm #1 (v > 0)
      v = fmaxf(fast_pow_pos(v, p), EPSV);         // power-norm #2
      __bf16 h, l;
      splitf(v, h, l);
      fh[m * DD + n] = h;
      fl[m * DD + n] = l;
    }
}

// ---------------------------------------------------------------------------
// Kernel 3a: one-shot split of W into bf16 hi/lo planes (4.2M elements)
// ---------------------------------------------------------------------------
__global__ __launch_bounds__(256) void wsplit_kernel(const float* __restrict__ W,
                                                     __bf16* __restrict__ Wh,
                                                     __bf16* __restrict__ Wl,
                                                     int n) {
  int i = blockIdx.x * 256 + threadIdx.x;
  if (i < n) {
    __bf16 h, l;
    splitf(W[i], h, l);
    Wh[i] = h; Wl[i] = l;
  }
}

// ---------------------------------------------------------------------------
// Kernel 3b: fc = flat[64,16384] @ W^T[16384,256] + b (one 16x16 tile/wave),
// reading pre-split bf16 planes -> zero conversion VALU in the hot loop.
// ---------------------------------------------------------------------------
__global__ __launch_bounds__(256) void fc_kernel(const __bf16* __restrict__ flatH,
                                                 const __bf16* __restrict__ flatL,
                                                 const __bf16* __restrict__ Wh,
                                                 const __bf16* __restrict__ Wl,
                                                 const float* __restrict__ bias,
                                                 float* __restrict__ fc) {
  int tid = threadIdx.x;
  int wave = tid >> 5;
  int lane = tid & 31, hl = lane >> 4, n0 = lane & 15;
  int tile = blockIdx.x * 8 + wave;  // 64 tiles = 4 row-tiles x 16 col-tiles
  int rt = tile >> 4;
  int ct = tile & 15;

  v8f acc;
#pragma unroll
  for (int r = 0; r < 8; ++r) acc[r] = 0.0f;

#pragma unroll 1
  for (int k0 = 0; k0 < KFC; k0 += 32) {
    __builtin_prefetch(Wh + (size_t)(ct * 16 + n0) * KFC + k0 + 256, 0, 1);
    v16bf ah = load_a_plane(flatH, KFC, rt * 16, k0);
    v16bf al = load_a_plane(flatL, KFC, rt * 16, k0);
    v16bf bh = load_b_rowk(Wh, KFC, k0, ct * 16);
    v16bf bl = load_b_rowk(Wl, KFC, k0, ct * 16);
    acc = wmma_bf16(ah, bh, acc);
    acc = wmma_bf16(ah, bl, acc);
    acc = wmma_bf16(al, bh, acc);
  }

#pragma unroll
  for (int r = 0; r < 8; ++r) {
    int m = rt * 16 + r + 8 * hl;
    int n = ct * 16 + n0;
    fc[m * OUTF + n] = acc[r] + bias[n];
  }
}

// ---------------------------------------------------------------------------
// Kernel 4: row-wise L2 normalize (64 rows x 256 cols)
// ---------------------------------------------------------------------------
__global__ __launch_bounds__(256) void l2norm_kernel(const float* __restrict__ fc,
                                                     float* __restrict__ out) {
  __shared__ float wsum[8];
  int row = blockIdx.x;
  int tid = threadIdx.x;
  float v = fc[row * OUTF + tid];
  float ss = v * v;
#pragma unroll
  for (int off = 16; off > 0; off >>= 1) ss += __shfl_xor(ss, off, 32);
  if ((tid & 31) == 0) wsum[tid >> 5] = ss;
  __syncthreads();
  if (tid == 0) {
    float s = 0.0f;
    for (int i = 0; i < 8; ++i) s += wsum[i];
    wsum[0] = s;
  }
  __syncthreads();
  float nrm = sqrtf(wsum[0]);
  out[row * OUTF + tid] = v / fmaxf(nrm, 1e-12f);
}

// ---------------------------------------------------------------------------
extern "C" void kernel_launch(void* const* d_in, const int* in_sizes, int n_in,
                              void* d_out, int out_size, void* d_ws, size_t ws_size,
                              hipStream_t stream) {
  const float* x    = (const float*)d_in[0];  // [64,4096,128]
  const float* W    = (const float*)d_in[1];  // [256,16384]
  const float* bias = (const float*)d_in[2];  // [256]
  const float* p    = (const float*)d_in[3];  // [1]
  float* out = (float*)d_out;                 // [64,256]

  const size_t MB = 1024 * 1024;
  char* w = (char*)d_ws;
  float*  cov   = (float*)(w);              // 4 MB  [64,128,128] f32
  __bf16* flatH = (__bf16*)(w + 4 * MB);    // 2 MB  [64,16384] bf16 hi
  __bf16* flatL = (__bf16*)(w + 6 * MB);    // 2 MB  [64,16384] bf16 lo
  __bf16* Wh    = (__bf16*)(w + 8 * MB);    // 8 MB  [256,16384] bf16 hi
  __bf16* Wl    = (__bf16*)(w + 16 * MB);   // 8 MB  [256,16384] bf16 lo
  float*  fc    = (float*)(w + 24 * MB);    // 64 KB [64,256] f32

  wsplit_kernel<<<(OUTF * KFC + 255) / 256, 256, 0, stream>>>(W, Wh, Wl,
                                                              OUTF * KFC);
  cov_kernel   <<<B_BATCH, 256, 0, stream>>>(x, cov);
  logm_kernel  <<<B_BATCH, 256, 0, stream>>>(cov, p, flatH, flatL);
  fc_kernel    <<<8,       256, 0, stream>>>(flatH, flatL, Wh, Wl, bias, fc);
  l2norm_kernel<<<B_BATCH, 256, 0, stream>>>(fc, out);

  (void)in_sizes; (void)n_in; (void)out_size; (void)ws_size;
}
